// MultiheadAttentionNLP_15221364097693
// MI455X (gfx1250) — matrix operand order repaired and checked by reading, not compiled, not run on hardware
//
#include <hip/hip_runtime.h>
#include <hip/hip_bf16.h>

typedef __attribute__((ext_vector_type(16))) __bf16 v16bf;
typedef __attribute__((ext_vector_type(8)))  float  v8f;

#if __has_builtin(__builtin_amdgcn_global_load_async_to_lds_b128) && \
    __has_builtin(__builtin_amdgcn_s_wait_asynccnt)
#define HAVE_ASYNC_LDS 1
#else
#define HAVE_ASYNC_LDS 0
#endif

namespace {
constexpr int kB = 8, kS = 2048, kF = 512, kM = 64, kHM = 512;
constexpr float kScale = 0.125f;      // rsqrt(M=64)
constexpr float kNeg   = -1.0e9f;

// ---- workspace layout (bytes) ----
constexpr size_t kXbBytes = (size_t)kB * kS * kF * 2;       // x in bf16, 16 MiB
constexpr size_t kWtBytes = (size_t)kF * kHM * 2;           // one transposed weight, 512 KiB
constexpr size_t kQKBytes = (size_t)8 * kB * kS * kM * 2;   // q / k / vT / msgs, 16 MiB each
constexpr size_t offXb   = 0;
constexpr size_t offWqT  = offXb  + kXbBytes;
constexpr size_t offWkT  = offWqT + kWtBytes;
constexpr size_t offWvT  = offWkT + kWtBytes;
constexpr size_t offWoT  = offWvT + kWtBytes;
constexpr size_t offQ    = offWoT + kWtBytes;
constexpr size_t offK    = offQ   + kQKBytes;
constexpr size_t offVT   = offK   + kQKBytes;
constexpr size_t offMsgs = offVT  + kQKBytes;
} // namespace

// ---------------- WMMA helpers ----------------
__device__ __forceinline__ v8f wmma_bf16(v16bf a, v16bf b, v8f c) {
  // D = A(16x32 bf16) * B(32x16 bf16) + C(16x16 f32)
  return __builtin_amdgcn_wmma_f32_16x16x32_bf16(false, a, false, b, (short)0, c,
                                                 false, false);
}

union FragU {
  v16bf v;
  uint4 q[2];
  unsigned int u[8];
  __bf16 e[16];
};

// A-matrix 16x32 bf16 fragment from row-major [row][k] memory (stride in elems).
__device__ __forceinline__ v16bf load_afrag(const __bf16* base, int stride, int row,
                                            int kbase, int lane) {
  const int half = lane >> 4;
  const __bf16* p0 = base + (size_t)row * stride + kbase + half * 16;
  FragU f;
  f.q[0] = *(const uint4*)(p0);
  f.q[1] = *(const uint4*)(p0 + 8);
  return f.v;
}

// B-matrix 32x16 bf16 fragment from [col][k] memory (stride in elems).
__device__ __forceinline__ v16bf load_bfrag(const __bf16* base, int stride, int col,
                                            int kbase, int lane) {
  const int half = lane >> 4;
  const __bf16* p = base + (size_t)col * stride + kbase + half * 16;
  FragU f;
  f.q[0] = *(const uint4*)(p);
  f.q[1] = *(const uint4*)(p + 8);
  return f.v;
}

#if HAVE_ASYNC_LDS
// Async global -> LDS, 16 bytes per lane (tracked by ASYNCcnt).
// Builtin parameter types (from hipcc diagnostic): pointers to
// 'int __attribute__((vector_size(16)))' in AS1 (global) / AS3 (LDS).
typedef int v4i_g __attribute__((vector_size(16)));
typedef __attribute__((address_space(1))) v4i_g* gptr_b128;
typedef __attribute__((address_space(3))) v4i_g* lptr_b128;
__device__ __forceinline__ void async_copy_b128(const void* g, void* l) {
  __builtin_amdgcn_global_load_async_to_lds_b128((gptr_b128)g, (lptr_b128)l, 0, 0);
}
#endif

// ---------------- Kernel 0a: x (f32) -> bf16 ----------------
__global__ __launch_bounds__(256) void cvt_x(const float* __restrict__ x,
                                             __bf16* __restrict__ xb) {
  int i = blockIdx.x * 256 + threadIdx.x;       // one lane = 8 elements
  const float4* p = (const float4*)x + (size_t)i * 2;
  float4 a = p[0], c = p[1];
  union { __bf16 e[8]; uint4 q; } o;
  o.e[0] = (__bf16)a.x; o.e[1] = (__bf16)a.y; o.e[2] = (__bf16)a.z; o.e[3] = (__bf16)a.w;
  o.e[4] = (__bf16)c.x; o.e[5] = (__bf16)c.y; o.e[6] = (__bf16)c.z; o.e[7] = (__bf16)c.w;
  ((uint4*)xb)[i] = o.q;
}

// ---------------- Kernel 0b: W[f][c] f32 -> WT[c][f] bf16 ----------------
__global__ __launch_bounds__(256) void transpose_w(
    const float* __restrict__ w0, const float* __restrict__ w1,
    const float* __restrict__ w2, const float* __restrict__ w3,
    __bf16* __restrict__ t0, __bf16* __restrict__ t1,
    __bf16* __restrict__ t2, __bf16* __restrict__ t3) {
  const float* src = (blockIdx.z == 0) ? w0 : (blockIdx.z == 1) ? w1
                   : (blockIdx.z == 2) ? w2 : w3;
  __bf16* dst = (blockIdx.z == 0) ? t0 : (blockIdx.z == 1) ? t1
              : (blockIdx.z == 2) ? t2 : t3;
  int idx = blockIdx.x * 256 + threadIdx.x;   // 0 .. 512*512-1
  int c = idx & 511, f = idx >> 9;
  dst[(size_t)c * 512 + f] = (__bf16)src[(size_t)f * 512 + c];
}

// ---------------- Kernel 1: fused QKV projection ----------------
// z=0: q[h][b][s][m], z=1: k[h][b][s][m], z=2: vT[h][b][m][s]
__global__ __launch_bounds__(256) void qkv_proj(
    const __bf16* __restrict__ xb,
    const __bf16* __restrict__ wqt, const __bf16* __restrict__ wkt,
    const __bf16* __restrict__ wvt,
    const float* __restrict__ bq, const float* __restrict__ bk,
    const float* __restrict__ bv,
    __bf16* __restrict__ qo, __bf16* __restrict__ ko, __bf16* __restrict__ vto) {
  __shared__ __align__(16) unsigned short tr[8][256];

  const int z = blockIdx.z;
  const __bf16* wt = (z == 0) ? wqt : (z == 1) ? wkt : wvt;
  const float* bias = (z == 0) ? bq : (z == 1) ? bk : bv;
  __bf16* outp = (z == 0) ? qo : (z == 1) ? ko : vto;

  const int wave = threadIdx.x >> 5, lane = threadIdx.x & 31;
  const int ln = lane & 15, half = lane >> 4;
  const int g0 = blockIdx.x * 128 + wave * 16;   // flat row (b*S+s) base of tile
  const int c0 = blockIdx.y * 16;                // output column base
  const int hq = c0 >> 6;                        // head of this column tile

  v8f acc;
#pragma unroll
  for (int r = 0; r < 8; ++r) acc[r] = 0.0f;

#pragma unroll 4
  for (int ks = 0; ks < 16; ++ks) {
    v16bf a = load_afrag(xb, kF, g0 + ln, ks * 32, lane);
    v16bf bfr = load_bfrag(wt, kF, c0 + ln, ks * 32, lane);
    acc = wmma_bf16(a, bfr, acc);
  }
  const float bb = bias[c0 + ln];

  if (z < 2) {
    // LDS transpose: C layout (col per lane) -> row-major bf16 global store
#pragma unroll
    for (int r = 0; r < 8; ++r) {
      union { __bf16 h; unsigned short u; } cv;
      cv.h = (__bf16)(acc[r] + bb);
      tr[wave][(half * 8 + r) * 16 + ln] = cv.u;
    }
    __syncthreads();
    uint4 val = *(const uint4*)&tr[wave][ln * 16 + half * 8];
    int g = g0 + ln;
    int b = g >> 11, s = g & (kS - 1);
    size_t dst = (((size_t)(hq * kB + b) * kS + s) * kM) + (c0 & 63) + half * 8;
    *(uint4*)(outp + dst) = val;
  } else {
    // vT: lane owns column m = c0+ln, rows s consecutive -> contiguous store
    union { __bf16 e[8]; uint4 q; } o;
#pragma unroll
    for (int r = 0; r < 8; ++r) o.e[r] = (__bf16)(acc[r] + bb);
    int b = g0 >> 11, s0t = g0 & (kS - 1);
    int m = (c0 & 63) + ln;
    size_t dst = (((size_t)(hq * kB + b) * kM + m) * kS) + s0t + half * 8;
    *(uint4*)(outp + dst) = o.q;
  }
}

// ---------------- Kernel 2: flash attention (transposed tiles) ----------------
// Grid: (S/128, H*B). Each wave owns 16 queries; block stages K/VT via LDS with
// async double-buffering (ASYNCcnt) when the toolchain exposes the builtin.
__global__ __launch_bounds__(256) void attn(
    const __bf16* __restrict__ q, const __bf16* __restrict__ k,
    const __bf16* __restrict__ vt, const float* __restrict__ maskp,
    __bf16* __restrict__ msgs) {
#if HAVE_ASYNC_LDS
  __shared__ __align__(16) __bf16 ks_s[2][32 * 64];  // double-buffered, 8 KiB
  __shared__ __align__(16) __bf16 vs_s[2][64 * 32];  // double-buffered, 8 KiB
#else
  __shared__ __align__(16) __bf16 ks_s[1][32 * 64];
  __shared__ __align__(16) __bf16 vs_s[1][64 * 32];
#endif

  const int hb = blockIdx.y;
  const int h = hb >> 3, b = hb & 7;
  const __bf16* qb = q + (size_t)hb * kS * kM;
  const __bf16* kb = k + (size_t)hb * kS * kM;
  const __bf16* vtb = vt + (size_t)hb * kM * kS;

  const int wave = threadIdx.x >> 5, lane = threadIdx.x & 31;
  const int ln = lane & 15, half = lane >> 4;
  const int s0 = blockIdx.x * 128 + wave * 16;
  const int tid = threadIdx.x;
  const int vrow = tid >> 2, vch = tid & 3;  // vT tile: 64 rows x 4 chunks

  // Q^T B-fragments are invariant across key loop (K dim = d, 2 chunks of 32)
  v16bf qf0 = load_bfrag(qb, kM, s0 + ln, 0, lane);
  v16bf qf1 = load_bfrag(qb, kM, s0 + ln, 32, lane);

  v8f acc[4];
#pragma unroll
  for (int d = 0; d < 4; ++d)
#pragma unroll
    for (int r = 0; r < 8; ++r) acc[d][r] = 0.0f;
  float m_run = -3.0e38f, l_run = 0.0f;

#if HAVE_ASYNC_LDS
  int buf = 0;
  // prologue: stage tile t0=0 into buffer 0
  async_copy_b128(kb + tid * 8, &ks_s[0][tid * 8]);
  async_copy_b128(vtb + (size_t)vrow * kS + vch * 8, &vs_s[0][vrow * 32 + vch * 8]);
  __builtin_amdgcn_s_wait_asynccnt(0);
  __syncthreads();
#endif

  for (int t0 = 0; t0 < kS; t0 += 32) {
#if HAVE_ASYNC_LDS
    const __bf16* ks_c = ks_s[buf];
    const __bf16* vs_c = vs_s[buf];
    if (t0 + 32 < kS) {  // kick off next tile into the other buffer
      const int nb = buf ^ 1;
      async_copy_b128(kb + (size_t)(t0 + 32) * kM + tid * 8, &ks_s[nb][tid * 8]);
      async_copy_b128(vtb + (size_t)vrow * kS + (t0 + 32) + vch * 8,
                      &vs_s[nb][vrow * 32 + vch * 8]);
    }
#else
    const __bf16* ks_c = ks_s[0];
    const __bf16* vs_c = vs_s[0];
    __syncthreads();  // previous-iteration readers done before overwrite
    ((uint4*)ks_s[0])[tid] = ((const uint4*)(kb + (size_t)t0 * kM))[tid];
    *(uint4*)&vs_s[0][vrow * 32 + vch * 8] =
        *(const uint4*)(vtb + (size_t)vrow * kS + t0 + vch * 8);
    if (t0 + 32 < kS)
      __builtin_prefetch(kb + (size_t)(t0 + 32) * kM + tid * 8, 0, 0);
    __syncthreads();
#endif

    // S^T tiles: rows = keys, cols = queries  (softmax dim lands per-lane)
    v8f st0, st1;
    {
      v8f c0t, c1t;
#pragma unroll
      for (int r = 0; r < 8; ++r) { c0t[r] = 0.0f; c1t[r] = 0.0f; }
      v16bf kf;
      kf = load_afrag(ks_c, kM, ln, 0, lane);       c0t = wmma_bf16(kf, qf0, c0t);
      kf = load_afrag(ks_c, kM, ln, 32, lane);      c0t = wmma_bf16(kf, qf1, c0t);
      kf = load_afrag(ks_c, kM, 16 + ln, 0, lane);  c1t = wmma_bf16(kf, qf0, c1t);
      kf = load_afrag(ks_c, kM, 16 + ln, 32, lane); c1t = wmma_bf16(kf, qf1, c1t);
      st0 = c0t; st1 = c1t;
    }

    // mask[b][s][t]: lane covers s = s0+ln, t = t0 + half*8 + r (contiguous)
    const float* mrow = maskp + ((size_t)b * kS + (s0 + ln)) * kS + t0 + half * 8;
    float4 ma0 = ((const float4*)mrow)[0];
    float4 ma1 = ((const float4*)mrow)[1];
    float4 mb0 = ((const float4*)(mrow + 16))[0];
    float4 mb1 = ((const float4*)(mrow + 16))[1];
    float mv0[8] = {ma0.x, ma0.y, ma0.z, ma0.w, ma1.x, ma1.y, ma1.z, ma1.w};
    float mv1[8] = {mb0.x, mb0.y, mb0.z, mb0.w, mb1.x, mb1.y, mb1.z, mb1.w};

    // online softmax over the 32 keys of this step
    float lg0[8], lg1[8];
    float tmax = -3.0e38f;
#pragma unroll
    for (int r = 0; r < 8; ++r) {
      lg0[r] = st0[r] * kScale + (1.0f - mv0[r]) * kNeg;
      lg1[r] = st1[r] * kScale + (1.0f - mv1[r]) * kNeg;
      tmax = fmaxf(tmax, fmaxf(lg0[r], lg1[r]));
    }
    tmax = fmaxf(tmax, __shfl_xor(tmax, 16, 32));  // combine lane halves
    float newm = fmaxf(m_run, tmax);
    float cs = __expf(m_run - newm);
    float p0[8], p1[8], rs = 0.0f;
#pragma unroll
    for (int r = 0; r < 8; ++r) {
      p0[r] = __expf(lg0[r] - newm);
      p1[r] = __expf(lg1[r] - newm);
      rs += p0[r] + p1[r];
    }
    rs += __shfl_xor(rs, 16, 32);
    l_run = l_run * cs + rs;
    m_run = newm;

    // build P^T B-fragment (K=32 keys, cols=queries); numerator uses p*mask
    union Pk { unsigned int u[4]; __bf16 e[8]; } own0, own1, oth0, oth1;
#pragma unroll
    for (int r = 0; r < 8; ++r) {
      own0.e[r] = (__bf16)(p0[r] * mv0[r]);
      own1.e[r] = (__bf16)(p1[r] * mv1[r]);
    }
#pragma unroll
    for (int i = 0; i < 4; ++i) {
      oth0.u[i] = __shfl_xor(own0.u[i], 16, 32);
      oth1.u[i] = __shfl_xor(own1.u[i], 16, 32);
    }
    FragU pf;
#pragma unroll
    for (int i = 0; i < 4; ++i) {
      pf.u[i]     = half ? oth1.u[i] : own0.u[i];   // K 0..7   (or 16..23)
      pf.u[i + 4] = half ? own1.u[i] : oth0.u[i];   // K 8..15  (or 24..31)
    }

    // O^T += V^T (16d x 32t) * P^T, with online rescale by cs
#pragma unroll
    for (int d = 0; d < 4; ++d) {
#pragma unroll
      for (int r = 0; r < 8; ++r) acc[d][r] *= cs;
      v16bf vf = load_afrag(vs_c, 32, d * 16 + ln, 0, lane);
      acc[d] = wmma_bf16(vf, pf.v, acc[d]);
    }

#if HAVE_ASYNC_LDS
    // next tile fully landed in LDS, and all waves done reading current tile
    __builtin_amdgcn_s_wait_asynccnt(0);
    __syncthreads();
    buf ^= 1;
#endif
  }

  // finalize: divide by l, store msgs[b][s][h*64 + d] (contiguous per lane)
  const float inv = 1.0f / l_run;
#pragma unroll
  for (int d = 0; d < 4; ++d) {
    union { __bf16 e[8]; uint4 q; } o;
#pragma unroll
    for (int r = 0; r < 8; ++r) o.e[r] = (__bf16)(acc[d][r] * inv);
    size_t dst = ((size_t)(b * kS + s0 + ln)) * kHM + h * kM + d * 16 + half * 8;
    *(uint4*)(msgs + dst) = o.q;
  }
}

// ---------------- Kernel 3: output projection ----------------
__global__ __launch_bounds__(256) void out_proj(
    const __bf16* __restrict__ msgs, const __bf16* __restrict__ wot,
    const float* __restrict__ bo, float* __restrict__ out) {
  __shared__ __align__(16) float trf[8][256];

  const int wave = threadIdx.x >> 5, lane = threadIdx.x & 31;
  const int ln = lane & 15, half = lane >> 4;
  const int g0 = blockIdx.x * 128 + wave * 16;
  const int o0 = blockIdx.y * 16;

  v8f acc;
#pragma unroll
  for (int r = 0; r < 8; ++r) acc[r] = 0.0f;

#pragma unroll 4
  for (int ks = 0; ks < 16; ++ks) {
    v16bf a = load_afrag(msgs, kHM, g0 + ln, ks * 32, lane);
    v16bf bfr = load_bfrag(wot, kHM, o0 + ln, ks * 32, lane);
    acc = wmma_bf16(a, bfr, acc);
  }
  const float bb = bo[o0 + ln];

#pragma unroll
  for (int r = 0; r < 8; ++r) trf[wave][(half * 8 + r) * 16 + ln] = acc[r] + bb;
  __syncthreads();
  float4 v0 = *(const float4*)&trf[wave][ln * 16 + half * 8];
  float4 v1 = *(const float4*)&trf[wave][ln * 16 + half * 8 + 4];
  size_t dst = (size_t)(g0 + ln) * kHM + o0 + half * 8;
  *(float4*)(out + dst) = v0;
  *(float4*)(out + dst + 4) = v1;
}

// ---------------- launch ----------------
extern "C" void kernel_launch(void* const* d_in, const int* /*in_sizes*/, int /*n_in*/,
                              void* d_out, int /*out_size*/, void* d_ws,
                              size_t /*ws_size*/, hipStream_t stream) {
  const float* x    = (const float*)d_in[0];
  const float* mask = (const float*)d_in[1];
  const float* Wq   = (const float*)d_in[2];
  const float* Wk   = (const float*)d_in[3];
  const float* Wv   = (const float*)d_in[4];
  const float* bq   = (const float*)d_in[5];
  const float* bk   = (const float*)d_in[6];
  const float* bv   = (const float*)d_in[7];
  const float* Wo   = (const float*)d_in[8];
  const float* bo   = (const float*)d_in[9];

  char* ws = (char*)d_ws;
  __bf16* xb   = (__bf16*)(ws + offXb);
  __bf16* wqt  = (__bf16*)(ws + offWqT);
  __bf16* wkt  = (__bf16*)(ws + offWkT);
  __bf16* wvt  = (__bf16*)(ws + offWvT);
  __bf16* wot  = (__bf16*)(ws + offWoT);
  __bf16* qw   = (__bf16*)(ws + offQ);
  __bf16* kw   = (__bf16*)(ws + offK);
  __bf16* vtw  = (__bf16*)(ws + offVT);
  __bf16* msgs = (__bf16*)(ws + offMsgs);

  cvt_x<<<dim3(4096), 256, 0, stream>>>(x, xb);                       // 8M elems / 8
  transpose_w<<<dim3(1024, 1, 4), 256, 0, stream>>>(Wq, Wk, Wv, Wo,
                                                    wqt, wkt, wvt, wot);
  qkv_proj<<<dim3(128, 32, 3), 256, 0, stream>>>(xb, wqt, wkt, wvt,
                                                 bq, bk, bv, qw, kw, vtw);
  attn<<<dim3(kS / 128, kB * 8), 256, 0, stream>>>(qw, kw, vtw, mask, msgs);
  out_proj<<<dim3(128, 32), 256, 0, stream>>>(msgs, wot, bo, (float*)d_out);
}
